// PointnetFPModule_17841294147729
// MI455X (gfx1250) — matrix operand order repaired and checked by reading, not compiled, not run on hardware
//
#include <hip/hip_runtime.h>
#include <hip/hip_bf16.h>

typedef __attribute__((ext_vector_type(16))) _Float16 v16h;
typedef __attribute__((ext_vector_type(8)))  _Float16 v8h;
typedef __attribute__((ext_vector_type(8)))  float    v8f;

#define B_    16
#define NPT   4096
#define MKN   1024
#define C1_   256
#define C2_   512
#define K1_   768   /* C2 + C1, concat order = [interpolated, unknow_feats] */
#define O1_   256
#define K2_   256
#define O2_   256
#define EPS_DIST 1e-8f
#define EPS_BN   1e-5f

// ---------------------------------------------------------------------------
// 1) three-NN: for each unknown point find 3 nearest known points (per batch)
// ---------------------------------------------------------------------------
__global__ __launch_bounds__(256) void three_nn_kernel(
    const float* __restrict__ unknown,  // (B, NPT, 3)
    const float* __restrict__ known,    // (B, MKN, 3)
    int* __restrict__ idx,              // (B, NPT, 3)
    float* __restrict__ wgt)            // (B, NPT, 3)
{
  __shared__ float kx[MKN], ky[MKN], kz[MKN];
  const int b   = blockIdx.y;
  const int tid = threadIdx.x;
  for (int r = 0; r < MKN / 256; ++r) {
    int j = tid + r * 256;
    const float* p = known + ((size_t)b * MKN + j) * 3;
    kx[j] = p[0]; ky[j] = p[1]; kz[j] = p[2];
  }
  __syncthreads();

  const int i = blockIdx.x * 256 + tid;
  const float* q = unknown + ((size_t)b * NPT + i) * 3;
  const float px = q[0], py = q[1], pz = q[2];

  float d0 = 3.4e38f, d1 = 3.4e38f, d2 = 3.4e38f;
  int   j0 = 0, j1 = 0, j2 = 0;
  for (int j = 0; j < MKN; ++j) {
    float dx = px - kx[j], dy = py - ky[j], dz = pz - kz[j];
    float d = dx * dx + dy * dy + dz * dz;
    if (d < d0)      { d2 = d1; j2 = j1; d1 = d0; j1 = j0; d0 = d; j0 = j; }
    else if (d < d1) { d2 = d1; j2 = j1; d1 = d;  j1 = j; }
    else if (d < d2) { d2 = d;  j2 = j; }
  }
  float r0 = 1.f / (sqrtf(fmaxf(d0, 0.f)) + EPS_DIST);
  float r1 = 1.f / (sqrtf(fmaxf(d1, 0.f)) + EPS_DIST);
  float r2 = 1.f / (sqrtf(fmaxf(d2, 0.f)) + EPS_DIST);
  float s  = r0 + r1 + r2;
  size_t base = ((size_t)b * NPT + i) * 3;
  idx[base + 0] = j0; idx[base + 1] = j1; idx[base + 2] = j2;
  wgt[base + 0] = r0 / s; wgt[base + 1] = r1 / s; wgt[base + 2] = r2 / s;
}

// ---------------------------------------------------------------------------
// 2) build X = concat(interp(known_feats), unknow_feats) in [b][i][c] f16
//    (K contiguous so GEMM LDS staging needs no transpose)
// ---------------------------------------------------------------------------
__global__ __launch_bounds__(256) void build_x_kernel(
    const int*   __restrict__ idx,          // (B, NPT, 3)
    const float* __restrict__ wgt,          // (B, NPT, 3)
    const float* __restrict__ known_feats,  // (B, C2, MKN)
    const float* __restrict__ unknow_feats, // (B, C1, NPT)
    _Float16*    __restrict__ X)            // (B, NPT, K1)
{
  const int b    = blockIdx.y;
  const int wave = threadIdx.x >> 5;
  const int lane = threadIdx.x & 31;
  const int i    = blockIdx.x * 8 + wave;

  size_t base = ((size_t)b * NPT + i) * 3;
  const int   j0 = idx[base], j1 = idx[base + 1], j2 = idx[base + 2];
  const float w0 = wgt[base], w1 = wgt[base + 1], w2 = wgt[base + 2];

  const float* kf = known_feats + (size_t)b * C2_ * MKN;
  _Float16* xo = X + ((size_t)b * NPT + i) * K1_;

  #pragma unroll
  for (int t = 0; t < C2_ / 32; ++t) {
    int c = lane + t * 32;
    const float* row = kf + (size_t)c * MKN;
    float v = w0 * row[j0] + w1 * row[j1] + w2 * row[j2];
    xo[c] = (_Float16)v;
  }
  const float* uf = unknow_feats + (size_t)b * C1_ * NPT;
  #pragma unroll
  for (int t = 0; t < C1_ / 32; ++t) {
    int c = lane + t * 32;
    xo[C2_ + c] = (_Float16)uf[(size_t)c * NPT + i];
  }
}

// ---------------------------------------------------------------------------
// 3) WMMA GEMM: Y[b] (O x N, f32) = A (O x K, f16) * X[b]^T (X stored N x K)
//    Block: 256 thr / 8 waves, 128(O) x 128(N) tile, K-step 32.
//    Tile staging: GLOBAL_LOAD_ASYNC_TO_LDS_B128 (ASYNCcnt), double-buffered
//    LDS so the next tile's DMA overlaps this tile's v_wmma work.
// ---------------------------------------------------------------------------
__global__ __launch_bounds__(256) void gemm_wmma_kernel(
    const _Float16* __restrict__ A,   // O x K row-major (weights, shared)
    const _Float16* __restrict__ Bm,  // per-batch N x K row-major (activations)
    float*          __restrict__ Y,   // per-batch O x N row-major
    int O, int K, int N)
{
  enum { SA = 40, SB = 40, BUF = 128 * SA };   // f16 elements per buffer
  __shared__ __align__(16) _Float16 As[2 * BUF];
  __shared__ __align__(16) _Float16 Bs[2 * BUF];

  const int b      = blockIdx.z;
  const int i0     = blockIdx.x * 128;       // N tile base
  const int o0     = blockIdx.y * 128;       // O tile base
  const int tid    = threadIdx.x;
  const int lane   = tid & 31;
  const int lane16 = lane & 15;
  const int upper  = lane >> 4;
  const int wave   = tid >> 5;
  const int wm     = wave & 1;               // O sub-tile (64 rows)
  const int wn     = wave >> 1;              // N sub-tile (32 cols)

  const _Float16* Bb = Bm + (size_t)b * (size_t)N * (size_t)K;
  float*          Yb = Y  + (size_t)b * (size_t)O * (size_t)N;

  const v8f vzero = {0.f, 0.f, 0.f, 0.f, 0.f, 0.f, 0.f, 0.f};
  v8f acc[4][2];
  #pragma unroll
  for (int mi = 0; mi < 4; ++mi)
    #pragma unroll
    for (int ni = 0; ni < 2; ++ni) acc[mi][ni] = vzero;

  const int rowT = tid >> 1;     // 0..127 : tile row this thread stages
  const int half = tid & 1;      // 0/1    : which 16-element K chunk

  // Async-stage one 128x32 A tile + 128x32 B tile into LDS buffer `buf`.
  // LDS dest operand is the 32-bit LDS byte offset (low 32 bits of the
  // generic LDS address per the aperture mapping); offset:16 applies to both
  // the global and LDS side per the ISA pseudocode.
  auto stage = [&](int k0, int buf) {
    unsigned long long gA = (unsigned long long)(uintptr_t)
        (A  + (size_t)(o0 + rowT) * K + k0 + half * 16);
    unsigned long long gB = (unsigned long long)(uintptr_t)
        (Bb + (size_t)(i0 + rowT) * K + k0 + half * 16);
    unsigned lA = (unsigned)(uintptr_t)(As + buf * BUF + rowT * SA + half * 16);
    unsigned lB = (unsigned)(uintptr_t)(Bs + buf * BUF + rowT * SB + half * 16);
    asm volatile(
        "global_load_async_to_lds_b128 %0, %2, off\n\t"
        "global_load_async_to_lds_b128 %0, %2, off offset:16\n\t"
        "global_load_async_to_lds_b128 %1, %3, off\n\t"
        "global_load_async_to_lds_b128 %1, %3, off offset:16"
        :
        : "v"(lA), "v"(lB), "v"(gA), "v"(gB)
        : "memory");
  };

  stage(0, 0);
  asm volatile("s_wait_asynccnt 0" ::: "memory");
  __syncthreads();

  int buf = 0;
  for (int k0 = 0; k0 < K; k0 += 32) {
    if (k0 + 32 < K) stage(k0 + 32, buf ^ 1);   // overlap DMA with WMMA

    // ---- fragments (ISA wave32 layouts, CDNA5 ISA §7.12.2) ----
    const _Float16* Ab = As + buf * BUF;
    const _Float16* Bq = Bs + buf * BUF;
    union Frag { v16h v; v8h h[2]; } af[4], bf[2];
    const int klo = upper ? 8  : 0;   // A: lanes16-31 hold K {8..15, 24..31}
    const int khi = upper ? 24 : 16;
    #pragma unroll
    for (int mi = 0; mi < 4; ++mi) {
      int r = wm * 64 + mi * 16 + lane16;
      af[mi].h[0] = *(const v8h*)(Ab + r * SA + klo);
      af[mi].h[1] = *(const v8h*)(Ab + r * SA + khi);
    }
    const int kb = upper ? 8 : 0;     // B: lanes16-31 hold K {8..15, 24..31} of column
    #pragma unroll
    for (int ni = 0; ni < 2; ++ni) {
      int c = wn * 32 + ni * 16 + lane16;
      bf[ni].h[0] = *(const v8h*)(Bq + c * SB + kb);
      bf[ni].h[1] = *(const v8h*)(Bq + c * SB + kb + 16);
    }

    #pragma unroll
    for (int mi = 0; mi < 4; ++mi)
      #pragma unroll
      for (int ni = 0; ni < 2; ++ni)
        acc[mi][ni] = __builtin_amdgcn_wmma_f32_16x16x32_f16(
            false, af[mi].v, false, bf[ni].v,
            (short)0, acc[mi][ni], false, false);

    asm volatile("s_wait_asynccnt 0" ::: "memory");  // own next-tile DMA done
    __syncthreads();                                 // whole block done
    buf ^= 1;
  }

  // ---- store: D layout lane16 = N col, VGPR v -> M = v + 8*upper ----
  #pragma unroll
  for (int mi = 0; mi < 4; ++mi) {
    int orow0 = o0 + wm * 64 + mi * 16 + upper * 8;
    #pragma unroll
    for (int ni = 0; ni < 2; ++ni) {
      int col = i0 + wn * 32 + ni * 16 + lane16;
      #pragma unroll
      for (int v = 0; v < 8; ++v)
        Yb[(size_t)(orow0 + v) * N + col] = acc[mi][ni][v];
    }
  }
}

// ---------------------------------------------------------------------------
// 4) BatchNorm statistics over (B, N) per channel (biased var, torch-style)
// ---------------------------------------------------------------------------
__global__ __launch_bounds__(256) void bn_stats_kernel(
    const float* __restrict__ Y, float* __restrict__ stats, int O, int N)
{
  const int o = blockIdx.x, tid = threadIdx.x;
  float s = 0.f, q = 0.f;
  for (int bb = 0; bb < B_; ++bb) {
    const float* row = Y + ((size_t)bb * O + o) * N;
    for (int ii = tid; ii < N; ii += 256) { float v = row[ii]; s += v; q += v * v; }
  }
  __shared__ float ss[256], sq[256];
  ss[tid] = s; sq[tid] = q; __syncthreads();
  for (int st = 128; st > 0; st >>= 1) {
    if (tid < st) { ss[tid] += ss[tid + st]; sq[tid] += sq[tid + st]; }
    __syncthreads();
  }
  if (tid == 0) {
    float cnt  = (float)B_ * (float)N;
    float mean = ss[0] / cnt;
    float var  = sq[0] / cnt - mean * mean;
    stats[o * 2]     = mean;
    stats[o * 2 + 1] = rsqrtf(var + EPS_BN);
  }
}

// ---------------------------------------------------------------------------
// 5) BN + ReLU, transposed f16 output [b][i][o] for next WMMA GEMM input
// ---------------------------------------------------------------------------
__global__ __launch_bounds__(256) void bn_apply_tr_kernel(
    const float* __restrict__ Y, const float* __restrict__ stats,
    const float* __restrict__ gamma, const float* __restrict__ beta,
    _Float16* __restrict__ H, int O, int N)
{
  __shared__ float tile[32][33];
  const int b  = blockIdx.z;
  const int o0 = blockIdx.y * 32, i0 = blockIdx.x * 32;
  const int t  = threadIdx.x;
  {
    int ic = t & 31, orow = t >> 5;
    #pragma unroll
    for (int r = 0; r < 4; ++r) {
      int ol = orow + r * 8, o = o0 + ol;
      float v = Y[((size_t)b * O + o) * N + i0 + ic];
      v = (v - stats[o * 2]) * stats[o * 2 + 1] * gamma[o] + beta[o];
      tile[ol][ic] = fmaxf(v, 0.f);
    }
  }
  __syncthreads();
  {
    int oc = t & 31, irow = t >> 5;
    #pragma unroll
    for (int r = 0; r < 4; ++r) {
      int il = irow + r * 8;
      H[((size_t)b * N + i0 + il) * O + o0 + oc] = (_Float16)tile[oc][il];
    }
  }
}

// ---------------------------------------------------------------------------
// 6) final BN + ReLU in place on d_out (f32, (B, O, N) layout)
// ---------------------------------------------------------------------------
__global__ __launch_bounds__(256) void bn_apply_inplace_kernel(
    float* __restrict__ Y, const float* __restrict__ stats,
    const float* __restrict__ gamma, const float* __restrict__ beta,
    int O, int N, size_t total)
{
  size_t t = (size_t)blockIdx.x * blockDim.x + threadIdx.x;
  if (t >= total) return;
  int o = (int)((t / (size_t)N) % (size_t)O);
  float v = Y[t];
  v = (v - stats[o * 2]) * stats[o * 2 + 1] * gamma[o] + beta[o];
  Y[t] = fmaxf(v, 0.f);
}

__global__ __launch_bounds__(256) void f32_to_f16_kernel(
    const float* __restrict__ src, _Float16* __restrict__ dst, int n)
{
  int t = blockIdx.x * blockDim.x + threadIdx.x;
  if (t < n) dst[t] = (_Float16)src[t];
}

// ---------------------------------------------------------------------------
extern "C" void kernel_launch(void* const* d_in, const int* in_sizes, int n_in,
                              void* d_out, int out_size, void* d_ws, size_t ws_size,
                              hipStream_t stream)
{
  (void)in_sizes; (void)n_in; (void)out_size; (void)ws_size;
  const float* unknown      = (const float*)d_in[0];
  const float* known        = (const float*)d_in[1];
  const float* unknow_feats = (const float*)d_in[2];
  const float* known_feats  = (const float*)d_in[3];
  const float* W1 = (const float*)d_in[4];
  const float* g1 = (const float*)d_in[5];
  const float* b1 = (const float*)d_in[6];
  const float* W2 = (const float*)d_in[7];
  const float* g2 = (const float*)d_in[8];
  const float* b2 = (const float*)d_in[9];
  float* out = (float*)d_out;

  char* p = (char*)d_ws;
  auto alloc = [&](size_t bytes) -> char* {
    char* r = p; p += (bytes + 255) & ~(size_t)255; return r;
  };
  int*      idx = (int*)      alloc((size_t)B_ * NPT * 3 * sizeof(int));
  float*    wgt = (float*)    alloc((size_t)B_ * NPT * 3 * sizeof(float));
  _Float16* W1h = (_Float16*) alloc((size_t)O1_ * K1_ * sizeof(_Float16));
  _Float16* W2h = (_Float16*) alloc((size_t)O2_ * K2_ * sizeof(_Float16));
  float*    st1 = (float*)    alloc((size_t)O1_ * 2 * sizeof(float));
  float*    st2 = (float*)    alloc((size_t)O2_ * 2 * sizeof(float));
  _Float16* X   = (_Float16*) alloc((size_t)B_ * NPT * K1_ * sizeof(_Float16));
  _Float16* H1  = (_Float16*) alloc((size_t)B_ * NPT * K2_ * sizeof(_Float16));
  float*    Y1  = (float*)    alloc((size_t)B_ * O1_ * NPT * sizeof(float));

  f32_to_f16_kernel<<<(O1_ * K1_ + 255) / 256, 256, 0, stream>>>(W1, W1h, O1_ * K1_);
  f32_to_f16_kernel<<<(O2_ * K2_ + 255) / 256, 256, 0, stream>>>(W2, W2h, O2_ * K2_);

  three_nn_kernel<<<dim3(NPT / 256, B_), 256, 0, stream>>>(unknown, known, idx, wgt);
  build_x_kernel<<<dim3(NPT / 8, B_), 256, 0, stream>>>(idx, wgt, known_feats,
                                                        unknow_feats, X);

  gemm_wmma_kernel<<<dim3(NPT / 128, O1_ / 128, B_), 256, 0, stream>>>(
      W1h, X, Y1, O1_, K1_, NPT);
  bn_stats_kernel<<<O1_, 256, 0, stream>>>(Y1, st1, O1_, NPT);
  bn_apply_tr_kernel<<<dim3(NPT / 32, O1_ / 32, B_), 256, 0, stream>>>(
      Y1, st1, g1, b1, H1, O1_, NPT);

  gemm_wmma_kernel<<<dim3(NPT / 128, O2_ / 128, B_), 256, 0, stream>>>(
      W2h, H1, out, O2_, K2_, NPT);
  bn_stats_kernel<<<O2_, 256, 0, stream>>>(out, st2, O2_, NPT);
  size_t total = (size_t)B_ * O2_ * NPT;
  bn_apply_inplace_kernel<<<(unsigned)((total + 255) / 256), 256, 0, stream>>>(
      out, st2, g2, b2, O2_, NPT, total);
}
